// ScoreNormCorrectedMetric_63453846831508
// MI455X (gfx1250) — matrix operand order repaired and checked by reading, not compiled, not run on hardware
//
#include <hip/hip_runtime.h>

// ---------------------------------------------------------------------------
// ScoreNormCorrectedMetric, MI455X / gfx1250 (wave32, WMMA).
//
//   s = x @ W^T                     (16384x2048 · 2048x2048)
//   out[b] = ||xd_b||^2 - (s_b·xd_b)^2 / (1 + ||s_b||^2)
//
// bf16x2 split GEMM on v_wmma_f32_16x16x32_bf16 (hi*hi + lo*hi + hi*lo =>
// ~fp32 accuracy), rank-1 collapse fused so s never touches HBM. W is
// pre-split into bf16 hi/lo planes in d_ws (requires ws_size >= 16 MiB).
//
// v2: software-pipelined B-tile loads (double buffer) + pipelined A loads
// across k, and 16 waves x 8 C-tiles (instead of 8 x 16) to halve
// accumulator pressure and raise occupancy — removes the
// "s_wait_loadcnt 0x0 before every WMMA pair" serialization seen in v1.
// ---------------------------------------------------------------------------

typedef __attribute__((ext_vector_type(16))) __bf16 v16bf;
typedef __attribute__((ext_vector_type(4)))  __bf16 v4bf;
typedef __attribute__((ext_vector_type(8)))  float  v8f;
typedef __attribute__((ext_vector_type(4)))  float  v4f;

#define D_DIM 2048
#define B_DIM 16384
#define NWAVE 16           // waves per workgroup
#define NT    8            // 16-col n-tiles per wave  (NWAVE*NT*16 == D_DIM)

static __device__ __forceinline__ v8f wmma_bf16(v16bf a, v16bf b, v8f c) {
    // (neg_a, A, neg_b, B, c_mod, C, reuse_a, reuse_b)
    return __builtin_amdgcn_wmma_f32_16x16x32_bf16(false, a, false, b,
                                                   (short)0, c, false, false);
}

// ---------------------------------------------------------------------------
// Kernel 1: split W (f32, row-major [e,d]) into bf16 hi + exact-residual lo.
// ---------------------------------------------------------------------------
__global__ __launch_bounds__(256) void split_w_kernel(
    const float* __restrict__ W, __bf16* __restrict__ whi, __bf16* __restrict__ wlo)
{
    const size_t i = ((size_t)blockIdx.x * 256u + threadIdx.x) * 4u;
    v4f w = *(const v4f*)(W + i);
    v4bf hi, lo;
#pragma unroll
    for (int h = 0; h < 4; ++h) {
        float f   = w[h];
        __bf16 fh = (__bf16)f;            // RNE to bf16
        hi[h] = fh;
        lo[h] = (__bf16)(f - (float)fh);  // residual is exact in f32
    }
    *(v4bf*)(whi + i) = hi;
    *(v4bf*)(wlo + i) = lo;
}

// ---------------------------------------------------------------------------
// Kernel 2: fused bf16x2 GEMM + rank-1 metric collapse.
//   grid.x = B/16 (one WG per 16 rows), 512 threads = 16 waves.
//   wave w owns output columns [w*128, (w+1)*128) as 8 n-tiles of 16.
// ---------------------------------------------------------------------------
__global__ __launch_bounds__(512, 1) void fused_metric_kernel(
    const float*  __restrict__ x,
    const float*  __restrict__ xdot,
    const __bf16* __restrict__ whi,
    const __bf16* __restrict__ wlo,
    float*        __restrict__ out)
{
    __shared__ float s_norm[16], s_sx[16], s_xx[16];

    const int tid     = threadIdx.x;
    const int lane    = tid & 31;
    const int wave    = tid >> 5;
    const int nlane   = lane & 15;     // column / row slot within 16-group
    const int half    = lane >> 4;     // 0: lanes 0-15, 1: lanes 16-31
    const int rowbase = blockIdx.x * 16;
    const int colbase = wave * (NT * 16);

    if (tid < 16) { s_norm[tid] = 0.f; s_sx[tid] = 0.f; s_xx[tid] = 0.f; }
    __syncthreads();

    // A-matrix source row for this lane (ISA 16-bit A layout: lanes L and
    // L+16 both carry row M = L&15, at K offsets selected by `half`).
    const float* xrow = x + (size_t)(rowbase + nlane) * D_DIM;

    // B-tile base for this lane: W row (colbase+nlane), plus per-half K run.
    // Tile nt lives at + nt*16 rows; per k-step advance by 32 elements.
    const __bf16* wh0 = whi + (size_t)(colbase + nlane) * D_DIM + half * 16;
    const __bf16* wl0 = wlo + (size_t)(colbase + nlane) * D_DIM + half * 16;
    const size_t  tile_stride = (size_t)16 * D_DIM;   // 16 W-rows per n-tile

    // 8 persistent accumulators (16x16 f32 tiles), one per n-tile.
    v8f C[NT];
#pragma unroll
    for (int nt = 0; nt < NT; ++nt)
#pragma unroll
        for (int j = 0; j < 8; ++j) C[nt][j] = 0.f;

    // ---- pipelined A load for k = 0 ----
    const int a_half = half * 16;
    v8f a0 = *(const v8f*)(xrow + a_half);
    v8f a1 = *(const v8f*)(xrow + a_half + 8);

    for (int k = 0; k < D_DIM / 32; ++k) {
        // ---- convert current A chunk to bf16 hi/lo fragments ----
        v16bf ahi, alo;
#pragma unroll
        for (int h = 0; h < 8; ++h) {
            float f0 = a0[h]; __bf16 h0 = (__bf16)f0;
            ahi[h]     = h0; alo[h]     = (__bf16)(f0 - (float)h0);
            float f1 = a1[h]; __bf16 h1 = (__bf16)f1;
            ahi[h + 8] = h1; alo[h + 8] = (__bf16)(f1 - (float)h1);
        }
        // ---- issue A loads for k+1 now; they fly during the n-tile loop ----
        if (k < D_DIM / 32 - 1) {
            const int c0 = (k + 1) * 32 + a_half;
            a0 = *(const v8f*)(xrow + c0);
            a1 = *(const v8f*)(xrow + c0 + 8);
        }

        const __bf16* bh_base = wh0 + k * 32;
        const __bf16* bl_base = wl0 + k * 32;

        // ---- double-buffered B tiles: load nt+1 while computing nt ----
        v16bf bh = *(const v16bf*)(bh_base);
        v16bf bl = *(const v16bf*)(bl_base);
#pragma unroll
        for (int nt = 0; nt < NT; ++nt) {
            v16bf nbh, nbl;
            if (nt < NT - 1) {
                nbh = *(const v16bf*)(bh_base + (size_t)(nt + 1) * tile_stride);
                nbl = *(const v16bf*)(bl_base + (size_t)(nt + 1) * tile_stride);
            }
            C[nt] = wmma_bf16(ahi, bh, C[nt]);   // hi*hi
            C[nt] = wmma_bf16(alo, bh, C[nt]);   // lo*hi
            C[nt] = wmma_bf16(ahi, bl, C[nt]);   // hi*lo
            if (nt < NT - 1) { bh = nbh; bl = nbl; }
        }
    }

    // ---- fold: per-row sums of s^2, s*xdot, xdot^2 ----
    // C layout: VGPR j, lane L -> s[m, n], m = j + 8*half, n = lane&15.
    float accn[8], accs[8], accx[8];
#pragma unroll
    for (int j = 0; j < 8; ++j) { accn[j] = 0.f; accs[j] = 0.f; accx[j] = 0.f; }

#pragma unroll
    for (int nt = 0; nt < NT; ++nt) {
        const int col = colbase + nt * 16 + nlane;
#pragma unroll
        for (int j = 0; j < 8; ++j) {
            const int m = j + half * 8;
            float s  = C[nt][j];
            float xd = xdot[(size_t)(rowbase + m) * D_DIM + col];
            accn[j] += s * s;
            accs[j] += s * xd;
            accx[j] += xd * xd;
        }
    }

    // reduce across the 16 columns held by each 16-lane half, then combine
    // the 16 waves through LDS float atomics (ds_add_f32).
#pragma unroll
    for (int j = 0; j < 8; ++j) {
        float n = accn[j], sv = accs[j], xv = accx[j];
#pragma unroll
        for (int mask = 1; mask < 16; mask <<= 1) {
            n  += __shfl_xor(n,  mask, 32);
            sv += __shfl_xor(sv, mask, 32);
            xv += __shfl_xor(xv, mask, 32);
        }
        if (nlane == 0) {
            const int m = j + half * 8;
            atomicAdd(&s_norm[m], n);
            atomicAdd(&s_sx[m],   sv);
            atomicAdd(&s_xx[m],   xv);
        }
    }
    __syncthreads();

    if (tid < 16) {
        float ns = s_norm[tid], sx = s_sx[tid], xx = s_xx[tid];
        out[rowbase + tid] = xx - (sx * sx) / (1.0f + ns);
    }
}

// ---------------------------------------------------------------------------
// Host entry. Inputs: d_in[0]=x [B,D] f32, d_in[1]=x_dot [B,D] f32,
// d_in[2]=W [D,D] f32. Output: [B] f32. Uses 16 MiB of d_ws for split-W.
// ---------------------------------------------------------------------------
extern "C" void kernel_launch(void* const* d_in, const int* in_sizes, int n_in,
                              void* d_out, int out_size, void* d_ws, size_t ws_size,
                              hipStream_t stream) {
    (void)in_sizes; (void)n_in; (void)out_size; (void)ws_size;
    const float* x    = (const float*)d_in[0];
    const float* xdot = (const float*)d_in[1];
    const float* W    = (const float*)d_in[2];
    float* out        = (float*)d_out;

    __bf16* whi = (__bf16*)d_ws;
    __bf16* wlo = whi + (size_t)D_DIM * D_DIM;   // ws usage: 2*D*D*2 = 16 MiB

    split_w_kernel<<<(D_DIM * D_DIM) / (256 * 4), 256, 0, stream>>>(W, whi, wlo);

    fused_metric_kernel<<<B_DIM / 16, NWAVE * 32, 0, stream>>>(x, xdot, whi, wlo, out);
}
